// TalkingHeadAttention_21440476742379
// MI455X (gfx1250) — matrix operand order repaired and checked by reading, not compiled, not run on hardware
//
#include <hip/hip_runtime.h>
#include <stdint.h>

// ---------------------------------------------------------------------------
// Talking-Head Attention for MI455X (gfx1250), wave32 + WMMA bf16.
// Pipeline: [q/k/v projections: bf16 WMMA GEMM] -> [fused 2-pass attention
// with talking-heads mixes around softmax] -> [output projection GEMM whose
// bf16 A-tile is staged with gfx1250 async global->LDS copies].
// ---------------------------------------------------------------------------

typedef __attribute__((ext_vector_type(16))) __bf16 v16bf;
typedef __attribute__((ext_vector_type(8)))  float  v8f;

union FragBF {
    v16bf bf;
    uint4 q[2];
    unsigned short u[16];
};

__device__ __forceinline__ unsigned short f2bf(float f) {
    union { float f; unsigned u; } v; v.f = f;
    unsigned r = v.u + 0x7FFFu + ((v.u >> 16) & 1u);   // round-to-nearest-even
    return (unsigned short)(r >> 16);
}
__device__ __forceinline__ float bf2f(unsigned short h) {
    union { unsigned u; float f; } v; v.u = ((unsigned)h) << 16;
    return v.f;
}
__device__ __forceinline__ v8f zero8() {
    v8f z;
    #pragma unroll
    for (int i = 0; i < 8; ++i) z[i] = 0.f;
    return z;
}
__device__ __forceinline__ float wred_max(float v) {
    #pragma unroll
    for (int off = 16; off > 0; off >>= 1) v = fmaxf(v, __shfl_xor(v, off, 32));
    return v;
}
__device__ __forceinline__ float wred_sum(float v) {
    #pragma unroll
    for (int off = 16; off > 0; off >>= 1) v += __shfl_xor(v, off, 32);
    return v;
}

// gfx1250 async global->LDS copy (ASYNCcnt-tracked), per-lane 16B.
// On AMDGCN the low 32 bits of a generic pointer to __shared__ are the LDS
// byte offset, which is what the VDST operand of the async load wants.
__device__ __forceinline__ void async_copy_b128(void* lds_ptr, const void* gaddr) {
    unsigned lds_off = (unsigned)(uintptr_t)lds_ptr;
    asm volatile("global_load_async_to_lds_b128 %0, %1, off"
                 :: "v"(lds_off), "v"(gaddr)
                 : "memory");
}
__device__ __forceinline__ void wait_async() {
    asm volatile("s_wait_asynccnt 0x0" ::: "memory");
}

// ---------------------------------------------------------------------------
// Tiled GEMM:  out[m, o] = A[m, :] . W[o, :] + bias[o]
// A: [8192, 768] fp32 (ABF=false) or bf16 (ABF=true).  W: [768,768] fp32.
// MODE 0: store bf16 head-major [b, h, n, hd]        (Q, K)
// MODE 1: store bf16 head-transposed [b, h, hd, n]   (V^T)
// MODE 2: store fp32 flat [m, o]                     (output projection)
// Block = 4 waves, tile 64(M) x 64(N), K-step 32.
// ---------------------------------------------------------------------------
template <int MODE, bool ABF>
__global__ __launch_bounds__(128) void th_gemm_kernel(
    const void* __restrict__ Aptr,
    const float* __restrict__ W, const float* __restrict__ bias,
    unsigned short* __restrict__ out_bf, float* __restrict__ out_f32)
{
    const int m0   = blockIdx.x * 64;
    const int n0   = blockIdx.y * 64;
    const int lane = threadIdx.x & 31;
    const int wave = threadIdx.x >> 5;

    __shared__ __align__(16) unsigned short As[64 * 32];
    __shared__ __align__(16) unsigned short Bs[64 * 32];

    v8f acc[4];
    #pragma unroll
    for (int i = 0; i < 4; ++i) acc[i] = zero8();

    for (int k0 = 0; k0 < 768; k0 += 32) {
        // ---- stage W tile (fp32 -> bf16, vectorized float4 loads) ----
        #pragma unroll
        for (int t = threadIdx.x * 4; t < 64 * 32; t += 128 * 4) {
            const int r = t >> 5, c = t & 31;
            const float4 w4 = *(const float4*)&W[(n0 + r) * 768 + k0 + c];
            unsigned short* d = &Bs[t];
            d[0] = f2bf(w4.x); d[1] = f2bf(w4.y);
            d[2] = f2bf(w4.z); d[3] = f2bf(w4.w);
        }
        // ---- stage A tile ----
        if constexpr (ABF) {
            // bf16 source: pure byte copy -> async global->LDS engine
            const unsigned short* A16 = (const unsigned short*)Aptr;
            #pragma unroll
            for (int t = threadIdx.x * 8; t < 64 * 32; t += 128 * 8) {
                const int r = t >> 5, c = t & 31;
                async_copy_b128(&As[t], &A16[(m0 + r) * 768 + k0 + c]);
            }
            wait_async();
        } else {
            const float* A32 = (const float*)Aptr;
            #pragma unroll
            for (int t = threadIdx.x * 4; t < 64 * 32; t += 128 * 4) {
                const int r = t >> 5, c = t & 31;
                const float4 a4 = *(const float4*)&A32[(m0 + r) * 768 + k0 + c];
                unsigned short* d = &As[t];
                d[0] = f2bf(a4.x); d[1] = f2bf(a4.y);
                d[2] = f2bf(a4.z); d[3] = f2bf(a4.w);
            }
        }
        __syncthreads();

        // A fragment (16x32): lanes 0-15 K {0-7,16-23}, lanes 16-31 K {8-15,24-31}
        FragBF a;
        const int arow = wave * 16 + (lane & 15);
        const int c0   = (lane & 16) ? 8 : 0;
        a.q[0] = *(const uint4*)&As[arow * 32 + c0];
        a.q[1] = *(const uint4*)&As[arow * 32 + c0 + 16];

        #pragma unroll
        for (int nb = 0; nb < 4; ++nb) {
            // B fragment (32x16): lane = output col, lanes 0-15 K {0-7,16-23}, lanes 16-31 K {8-15,24-31}
            FragBF bm;
            const int ocol = nb * 16 + (lane & 15);
            const int kb0  = (lane & 16) ? 8 : 0;
            bm.q[0] = *(const uint4*)&Bs[ocol * 32 + kb0];
            bm.q[1] = *(const uint4*)&Bs[ocol * 32 + kb0 + 16];
            acc[nb] = __builtin_amdgcn_wmma_f32_16x16x32_bf16(
                false, a.bf, false, bm.bf, (short)0, acc[nb], false, false);
        }
        __syncthreads();
    }

    // C/D layout: VGPR r -> M = r + 8*(lane>=16), N = lane&15
    const int rbase = (lane & 16) ? 8 : 0;
    #pragma unroll
    for (int nb = 0; nb < 4; ++nb) {
        const int ocol = n0 + nb * 16 + (lane & 15);
        const float bv = bias[ocol];
        #pragma unroll
        for (int r = 0; r < 8; ++r) {
            const int gm = m0 + wave * 16 + rbase + r;
            const float v = acc[nb][r] + bv;
            if constexpr (MODE == 2) {
                out_f32[(size_t)gm * 768 + ocol] = v;
            } else {
                const int bb = gm >> 10, n = gm & 1023;
                const int h = ocol >> 6, d = ocol & 63;
                if constexpr (MODE == 0)
                    out_bf[(((size_t)bb * 12 + h) * 1024 + n) * 64 + d] = f2bf(v);
                else
                    out_bf[(((size_t)bb * 12 + h) * 64 + d) * 1024 + n] = f2bf(v);
            }
        }
    }
}

// ---------------------------------------------------------------------------
// One wave's QK^T for a 16(query) x 32(key) tile -> scaled logits into LDS.
// ---------------------------------------------------------------------------
__device__ __forceinline__ void compute_logits_tile(
    const unsigned short* __restrict__ K, const FragBF* qa,
    float* __restrict__ raw_g, int kt0, int lane)
{
    const int rbase = (lane & 16) ? 8 : 0;
    #pragma unroll
    for (int nt = 0; nt < 2; ++nt) {
        v8f c = zero8();
        const int kn = kt0 + nt * 16 + (lane & 15);       // key row for B frag
        #pragma unroll
        for (int half = 0; half < 2; ++half) {            // hd chunks 0-31, 32-63
            FragBF kf;
            const int kk0 = half * 32 + ((lane & 16) ? 8 : 0);
            kf.q[0] = *(const uint4*)&K[kn * 64 + kk0];
            kf.q[1] = *(const uint4*)&K[kn * 64 + kk0 + 16];
            c = __builtin_amdgcn_wmma_f32_16x16x32_bf16(
                false, qa[half].bf, false, kf.bf, (short)0, c, false, false);
        }
        const int col = nt * 16 + (lane & 15);
        #pragma unroll
        for (int r = 0; r < 8; ++r)
            raw_g[(rbase + r) * 32 + col] = c[r] * 0.125f;  // 64^-0.5
    }
}

// ---------------------------------------------------------------------------
// Fused talking-heads attention. Grid (N/16, B), 12 waves = 12 heads.
// Pass A: logits -> mix1 -> online softmax stats (per-row max/sum).
// Pass B: logits -> mix1 -> normalize -> mix2 -> bf16 -> P @ V via WMMA.
// ---------------------------------------------------------------------------
__global__ __launch_bounds__(384) void th_attn_kernel(
    const unsigned short* __restrict__ qb, const unsigned short* __restrict__ kb,
    const unsigned short* __restrict__ vt,
    const float* __restrict__ th1w, const float* __restrict__ th1b,
    const float* __restrict__ th2w, const float* __restrict__ th2b,
    unsigned short* __restrict__ ctx)
{
    const int b    = blockIdx.y;
    const int nq0  = blockIdx.x * 16;
    const int g    = threadIdx.x >> 5;   // wave == head
    const int lane = threadIdx.x & 31;

    __shared__ __align__(16) float          lds_raw[12 * 512];  // raw logits
    __shared__ __align__(16) float          lds_p  [12 * 512];  // normalized probs
    __shared__ __align__(16) unsigned short lds_a  [12 * 512];  // mix2 probs, bf16

    float t1[12], t2[12];
    #pragma unroll
    for (int h = 0; h < 12; ++h) { t1[h] = th1w[g * 12 + h]; t2[h] = th2w[g * 12 + h]; }
    const float t1b = th1b[g], t2b = th2b[g];

    const unsigned short* Q = qb + (size_t)(b * 12 + g) * 1024 * 64;
    const unsigned short* K = kb + (size_t)(b * 12 + g) * 1024 * 64;
    const unsigned short* V = vt + (size_t)(b * 12 + g) * 64 * 1024;

    const int c0    = (lane & 16) ? 8 : 0;
    const int rbase = (lane & 16) ? 8 : 0;
    const int qm    = nq0 + (lane & 15);

    FragBF qa[2];
    #pragma unroll
    for (int half = 0; half < 2; ++half) {
        qa[half].q[0] = *(const uint4*)&Q[qm * 64 + half * 32 + c0];
        qa[half].q[1] = *(const uint4*)&Q[qm * 64 + half * 32 + c0 + 16];
    }

    float mrun[16], lrun[16];
    #pragma unroll
    for (int i = 0; i < 16; ++i) { mrun[i] = -3.0e38f; lrun[i] = 0.f; }

    // ------------------ pass A: softmax stats after mix1 ------------------
    #pragma unroll 1
    for (int kt0 = 0; kt0 < 1024; kt0 += 32) {
        if (kt0 + 32 < 1024) __builtin_prefetch(K + (size_t)(kt0 + 32) * 64, 0, 1);
        compute_logits_tile(K, qa, &lds_raw[g * 512], kt0, lane);
        __syncthreads();

        // lane holds column k=lane of each of the 16 rows for head g after mix1
        #pragma unroll
        for (int i = 0; i < 16; ++i) {
            float s = t1b;
            #pragma unroll
            for (int h = 0; h < 12; ++h) s += t1[h] * lds_raw[h * 512 + i * 32 + lane];
            const float tmax = wred_max(s);
            const float mn   = fmaxf(mrun[i], tmax);
            const float ssum = wred_sum(__expf(s - mn));
            lrun[i] = lrun[i] * __expf(mrun[i] - mn) + ssum;
            mrun[i] = mn;
        }
        __syncthreads();
    }

    // hoist softmax reciprocal out of the pass-B loop
    float rinv[16];
    #pragma unroll
    for (int i = 0; i < 16; ++i) rinv[i] = 1.0f / lrun[i];

    // ------------- pass B: normalize, mix2, accumulate P @ V --------------
    v8f oacc[4];
    #pragma unroll
    for (int i = 0; i < 4; ++i) oacc[i] = zero8();

    #pragma unroll 1
    for (int kt0 = 0; kt0 < 1024; kt0 += 32) {
        if (kt0 + 32 < 1024) __builtin_prefetch(V + (size_t)(kt0 + 32), 0, 1);
        compute_logits_tile(K, qa, &lds_raw[g * 512], kt0, lane);
        __syncthreads();

        #pragma unroll
        for (int i = 0; i < 16; ++i) {
            float s = t1b;
            #pragma unroll
            for (int h = 0; h < 12; ++h) s += t1[h] * lds_raw[h * 512 + i * 32 + lane];
            lds_p[g * 512 + i * 32 + lane] = __expf(s - mrun[i]) * rinv[i];
        }
        __syncthreads();

        #pragma unroll
        for (int i = 0; i < 16; ++i) {
            float a2 = t2b;
            #pragma unroll
            for (int h = 0; h < 12; ++h) a2 += t2[h] * lds_p[h * 512 + i * 32 + lane];
            lds_a[g * 512 + i * 32 + lane] = f2bf(a2);
        }
        __syncthreads();

        // A fragment of mixed probabilities (16 queries x 32 keys)
        FragBF pa;
        pa.q[0] = *(const uint4*)&lds_a[g * 512 + (lane & 15) * 32 + c0];
        pa.q[1] = *(const uint4*)&lds_a[g * 512 + (lane & 15) * 32 + c0 + 16];

        // V^T: row = hd, contiguous along keys -> clean 32 B B-fragment loads
        const int kr0 = kt0 + ((lane & 16) ? 8 : 0);
        #pragma unroll
        for (int nb = 0; nb < 4; ++nb) {
            FragBF vf;
            const int dcol = nb * 16 + (lane & 15);
            vf.q[0] = *(const uint4*)&V[dcol * 1024 + kr0];
            vf.q[1] = *(const uint4*)&V[dcol * 1024 + kr0 + 16];
            oacc[nb] = __builtin_amdgcn_wmma_f32_16x16x32_bf16(
                false, pa.bf, false, vf.bf, (short)0, oacc[nb], false, false);
        }
        __syncthreads();
    }

    // store context [b, n, h*64 + d] as bf16 for the output projection
    #pragma unroll
    for (int nb = 0; nb < 4; ++nb) {
        const int dglob = g * 64 + nb * 16 + (lane & 15);
        #pragma unroll
        for (int r = 0; r < 8; ++r) {
            const int n = nq0 + rbase + r;
            ctx[((size_t)(b * 1024 + n)) * 768 + dglob] = f2bf(oacc[nb][r]);
        }
    }
}

// ---------------------------------------------------------------------------
extern "C" void kernel_launch(void* const* d_in, const int* in_sizes, int n_in,
                              void* d_out, int out_size, void* d_ws, size_t ws_size,
                              hipStream_t stream)
{
    const float* x    = (const float*)d_in[0];
    const float* wq   = (const float*)d_in[1];
    const float* bq   = (const float*)d_in[2];
    const float* wk   = (const float*)d_in[3];
    const float* bk   = (const float*)d_in[4];
    const float* wv   = (const float*)d_in[5];
    const float* bv   = (const float*)d_in[6];
    const float* wo   = (const float*)d_in[7];
    const float* bo   = (const float*)d_in[8];
    const float* th1w = (const float*)d_in[9];
    const float* th1b = (const float*)d_in[10];
    const float* th2w = (const float*)d_in[11];
    const float* th2b = (const float*)d_in[12];

    const size_t SZ = (size_t)8 * 12 * 1024 * 64 * sizeof(unsigned short); // 12.58 MB
    uint8_t* ws = (uint8_t*)d_ws;
    unsigned short* qb  = (unsigned short*)(ws);
    unsigned short* kb  = (unsigned short*)(ws + SZ);
    unsigned short* vt  = (unsigned short*)(ws + 2 * SZ);
    unsigned short* ctx = (unsigned short*)(ws + 3 * SZ);

    dim3 ggrid(128, 12), gblk(128);
    // projections: Q, K head-major; V stored transposed per head
    th_gemm_kernel<0, false><<<ggrid, gblk, 0, stream>>>(x, wq, bq, qb, nullptr);
    th_gemm_kernel<0, false><<<ggrid, gblk, 0, stream>>>(x, wk, bk, kb, nullptr);
    th_gemm_kernel<1, false><<<ggrid, gblk, 0, stream>>>(x, wv, bv, vt, nullptr);

    // fused talking-heads attention
    th_attn_kernel<<<dim3(64, 8), 384, 0, stream>>>(qb, kb, vt,
                                                    th1w, th1b, th2w, th2b, ctx);

    // output projection -> fp32 (bf16 A staged via async global->LDS)
    th_gemm_kernel<2, true><<<ggrid, gblk, 0, stream>>>(ctx, wo, bo, nullptr,
                                                        (float*)d_out);
}
